// GPT_70574902608166
// MI455X (gfx1250) — matrix operand order repaired and checked by reading, not compiled, not run on hardware
//
#include <hip/hip_runtime.h>

// ---------------------------------------------------------------------------
// GPT-2 small forward for MI455X (gfx1250, wave32, WMMA).
//  - Dense GEMMs: v_wmma_f32_16x16x32_bf16, vectorized fp32->bf16 LDS staging,
//    1-deep register pipeline.
//  - Attention: fused per (b,h,64-row) block. Scores (WMMA) -> raw write to
//    the d_out map region + exact softmax stats in LDS -> L2-hot re-read ->
//    normalized map write + y = P@V (WMMA). Maps touch HBM ~once instead of 5x.
// ---------------------------------------------------------------------------

typedef __bf16 bf16_t;
typedef __attribute__((ext_vector_type(4)))  float  v4f;
typedef __attribute__((ext_vector_type(8)))  __bf16 v8bf;
typedef __attribute__((ext_vector_type(16))) __bf16 v16bf;
typedef __attribute__((ext_vector_type(8)))  float  v8f;

#define BM 64
#define BN 64
#define BK 32
#define GEMM_THREADS 256
#define NEG_INF (-3.0e38f)

__device__ __forceinline__ v8bf pack_bf16(v4f a, v4f b) {
  v8bf o;
  o[0] = (bf16_t)a[0]; o[1] = (bf16_t)a[1]; o[2] = (bf16_t)a[2]; o[3] = (bf16_t)a[3];
  o[4] = (bf16_t)b[0]; o[5] = (bf16_t)b[1]; o[6] = (bf16_t)b[2]; o[7] = (bf16_t)b[3];
  return o;
}

// Concatenate two contiguous 8-half LDS reads into a 16-wide A fragment.
__device__ __forceinline__ v16bf load_afrag(const bf16_t* pa) {
  v8bf alo = *(const v8bf*)pa;
  v8bf ahi = *(const v8bf*)(pa + 16);
  return __builtin_shufflevector(alo, ahi,
      0, 1, 2, 3, 4, 5, 6, 7, 8, 9, 10, 11, 12, 13, 14, 15);
}

__device__ __forceinline__ float redmax16(float v) {
  v = fmaxf(v, __shfl_xor(v, 1, 32));
  v = fmaxf(v, __shfl_xor(v, 2, 32));
  v = fmaxf(v, __shfl_xor(v, 4, 32));
  v = fmaxf(v, __shfl_xor(v, 8, 32));
  return v;
}
__device__ __forceinline__ float redsum16(float v) {
  v += __shfl_xor(v, 1, 32);
  v += __shfl_xor(v, 2, 32);
  v += __shfl_xor(v, 4, 32);
  v += __shfl_xor(v, 8, 32);
  return v;
}

// ---------------------------------------------------------------------------
// Generic GEMM: C[m,n] (+)= sum_k A[m,k] * W[n,k] (+ bias[n])   (C = A @ W^T)
// K must be a multiple of BK (768/3072 here). GUARD handles ragged M/N
// (logits GEMM only), with per-row guards.
// ---------------------------------------------------------------------------
template <bool HAS_BIAS, bool ACCUM, bool GUARD>
__global__ __launch_bounds__(GEMM_THREADS)
void gemm_bf16w(const float* __restrict__ A, const float* __restrict__ Bw,
                const float* __restrict__ bias, float* __restrict__ C,
                int M, int N, int K,
                long long lda, long long ldb, long long ldc) {
  const int n0 = blockIdx.x * BN;
  const int m0 = blockIdx.y * BM;

  __shared__ __align__(128) bf16_t sA[BM][BK];
  __shared__ __align__(128) bf16_t sB[BN][BK];

  const int tid     = threadIdx.x;
  const int wave    = tid >> 5;
  const int lane    = tid & 31;
  const int halfSel = lane >> 4;
  const int lr      = lane & 15;
  const int waveM   = wave >> 1;   // 0..3
  const int waveN   = wave & 1;    // 0..1

  v8f acc0 = {0.f, 0.f, 0.f, 0.f, 0.f, 0.f, 0.f, 0.f};
  v8f acc1 = {0.f, 0.f, 0.f, 0.f, 0.f, 0.f, 0.f, 0.f};

  const int aRow = tid >> 2;
  const int aK0  = (tid & 3) * 8;
  const float* aPtr = A  + (size_t)(m0 + aRow) * lda + aK0;
  const float* bPtr = Bw + (size_t)(n0 + aRow) * ldb + aK0;
  const bool aOK = !GUARD || (m0 + aRow) < M;
  const bool bOK = !GUARD || (n0 + aRow) < N;

  v4f aR0, aR1, bR0, bR1;
  {
    const v4f zz = {0.f, 0.f, 0.f, 0.f};
    if (aOK) { aR0 = *(const v4f*)aPtr; aR1 = *(const v4f*)(aPtr + 4); }
    else     { aR0 = zz; aR1 = zz; }
    if (bOK) { bR0 = *(const v4f*)bPtr; bR1 = *(const v4f*)(bPtr + 4); }
    else     { bR0 = zz; bR1 = zz; }
  }

  for (int kt = 0; kt < K; kt += BK) {
    *(v8bf*)&sA[aRow][aK0] = pack_bf16(aR0, aR1);
    *(v8bf*)&sB[aRow][aK0] = pack_bf16(bR0, bR1);
    __syncthreads();

    if (kt + BK < K) {
      const v4f zz = {0.f, 0.f, 0.f, 0.f};
      if (aOK) { aR0 = *(const v4f*)(aPtr + kt + BK);
                 aR1 = *(const v4f*)(aPtr + kt + BK + 4); }
      else     { aR0 = zz; aR1 = zz; }
      if (bOK) { bR0 = *(const v4f*)(bPtr + kt + BK);
                 bR1 = *(const v4f*)(bPtr + kt + BK + 4); }
      else     { bR0 = zz; bR1 = zz; }
    }

    v16bf afrag  = *(const v16bf*)&sA[waveM * 16 + lr][halfSel * 16];
    v16bf bfrag0 = *(const v16bf*)&sB[waveN * 32 + lr][halfSel * 16];
    v16bf bfrag1 = *(const v16bf*)&sB[waveN * 32 + 16 + lr][halfSel * 16];

    acc0 = __builtin_amdgcn_wmma_f32_16x16x32_bf16(false, afrag, false, bfrag0,
                                                   (short)0, acc0, false, false);
    acc1 = __builtin_amdgcn_wmma_f32_16x16x32_bf16(false, afrag, false, bfrag1,
                                                   (short)0, acc1, false, false);
    __syncthreads();
  }

  const int c0 = n0 + waveN * 32 + lr;
  const int c1 = c0 + 16;
  const int rbase = m0 + waveM * 16 + halfSel * 8;
  const float bias0 = (HAS_BIAS && (!GUARD || c0 < N)) ? bias[c0] : 0.0f;
  const float bias1 = (HAS_BIAS && (!GUARD || c1 < N)) ? bias[c1] : 0.0f;
  #pragma unroll
  for (int i = 0; i < 8; ++i) {
    const int r = rbase + i;
    if (GUARD && r >= M) continue;
    if (!GUARD || c0 < N) {
      float v = acc0[i] + bias0;
      if (ACCUM) C[(size_t)r * ldc + c0] += v;
      else       C[(size_t)r * ldc + c0]  = v;
    }
    if (!GUARD || c1 < N) {
      float v = acc1[i] + bias1;
      if (ACCUM) C[(size_t)r * ldc + c1] += v;
      else       C[(size_t)r * ldc + c1]  = v;
    }
  }
}

// ---------------------------------------------------------------------------
// Fused causal attention for one (b, h, 64-query-row) block.
//   P1: for each live 64-key tile: S = scale * q @ k^T (WMMA), write raw
//       scaled scores to the map region, record per-tile row max / expsum.
//   P2: exact row max & sum from per-tile stats (LDS only).
//   P3: re-read own score strip (L2-hot), write softmax probabilities
//       (zeros above diagonal), and accumulate y = P @ V via WMMA.
// D = 64, tiles are 64x64, 256 threads = 8 waves (4 row-strips x 2 col-strips).
// ---------------------------------------------------------------------------
__global__ __launch_bounds__(256)
void attn_fused_kernel(const float* __restrict__ qkv, float* __restrict__ att,
                       float* __restrict__ y, int T, int E3, int E, int Hn,
                       float scale) {
  const int z = blockIdx.z;
  const int b = z / Hn;
  const int h = z - b * Hn;
  const int mBlk = blockIdx.y;
  const int m0   = mBlk * 64;

  const float* qp = qkv + (size_t)b * T * E3 + (size_t)h * 64;
  const float* kp = qp + E;
  const float* vp = qp + 2 * E;
  float* attZ = att + (size_t)z * T * T;
  float* yp   = y + (size_t)b * T * E + (size_t)h * 64;

  __shared__ __align__(128) bf16_t sQ[64][64];
  __shared__ __align__(128) bf16_t sKV[64][64];
  __shared__ __align__(128) bf16_t sP[64][64];
  __shared__ float sTM[16][64];     // per-tile row max
  __shared__ float sTS[16][64];     // per-tile row expsum (rel. tile max)
  __shared__ float sPartM[64][2];
  __shared__ float sPartS[64][2];
  __shared__ float sMfin[64];
  __shared__ float sInv[64];

  const int tid     = threadIdx.x;
  const int wave    = tid >> 5;
  const int lane    = tid & 31;
  const int halfSel = lane >> 4;
  const int lr      = lane & 15;
  const int waveM   = wave >> 1;
  const int waveN   = wave & 1;
  const int rb      = waveM * 16 + halfSel * 8;   // local row base (0..56)
  const int cl0     = waveN * 32 + lr;            // local col 0
  const int cl1     = cl0 + 16;                   // local col 1

  // ---- stage Q block (rows m0..m0+63, D=64) ----
  #pragma unroll
  for (int cidx = 0; cidx < 2; ++cidx) {
    const int chunk = tid + cidx * 256;
    const int r = chunk >> 3, d0 = (chunk & 7) * 8;
    const float* p = qp + (size_t)(m0 + r) * E3 + d0;
    v4f x0 = *(const v4f*)p, x1 = *(const v4f*)(p + 4);
    *(v8bf*)&sQ[r][d0] = pack_bf16(x0, x1);
  }

  // ================= Phase 1: scores + stats =================
  for (int kn = 0; kn <= mBlk; ++kn) {
    const int n0t = kn * 64;
    // stage K tile [keyRow][d]
    #pragma unroll
    for (int cidx = 0; cidx < 2; ++cidx) {
      const int chunk = tid + cidx * 256;
      const int r = chunk >> 3, d0 = (chunk & 7) * 8;
      const float* p = kp + (size_t)(n0t + r) * E3 + d0;
      v4f x0 = *(const v4f*)p, x1 = *(const v4f*)(p + 4);
      *(v8bf*)&sKV[r][d0] = pack_bf16(x0, x1);
    }
    __syncthreads();

    v8f s0 = {0.f, 0.f, 0.f, 0.f, 0.f, 0.f, 0.f, 0.f};
    v8f s1 = {0.f, 0.f, 0.f, 0.f, 0.f, 0.f, 0.f, 0.f};
    #pragma unroll
    for (int ks = 0; ks < 2; ++ks) {
      v16bf af = *(const v16bf*)&sQ[waveM * 16 + lr][ks * 32 + halfSel * 16];
      v16bf bf0 = *(const v16bf*)&sKV[waveN * 32 + lr][ks * 32 + halfSel * 16];
      v16bf bf1 = *(const v16bf*)&sKV[waveN * 32 + 16 + lr][ks * 32 + halfSel * 16];
      s0 = __builtin_amdgcn_wmma_f32_16x16x32_bf16(false, af, false, bf0,
                                                   (short)0, s0, false, false);
      s1 = __builtin_amdgcn_wmma_f32_16x16x32_bf16(false, af, false, bf1,
                                                   (short)0, s1, false, false);
    }
    #pragma unroll
    for (int i = 0; i < 8; ++i) { s0[i] *= scale; s1[i] *= scale; }

    // raw scaled scores -> map region (masked positions overwritten in P3)
    const int cg0 = n0t + cl0, cg1 = n0t + cl1;
    #pragma unroll
    for (int i = 0; i < 8; ++i) {
      attZ[(size_t)(m0 + rb + i) * T + cg0] = s0[i];
      attZ[(size_t)(m0 + rb + i) * T + cg1] = s1[i];
    }

    // per-row tile max (causal-masked), merged across the two column waves
    float mloc[8];
    #pragma unroll
    for (int i = 0; i < 8; ++i) {
      const int rg = m0 + rb + i;
      float a = (cg0 <= rg) ? s0[i] : NEG_INF;
      float c = (cg1 <= rg) ? s1[i] : NEG_INF;
      mloc[i] = redmax16(fmaxf(a, c));
    }
    if (lr == 0) {
      #pragma unroll
      for (int i = 0; i < 8; ++i) sPartM[rb + i][waveN] = mloc[i];
    }
    __syncthreads();
    if (tid < 64) sTM[kn][tid] = fmaxf(sPartM[tid][0], sPartM[tid][1]);
    __syncthreads();

    // per-row tile expsum relative to tile max
    float eloc[8];
    #pragma unroll
    for (int i = 0; i < 8; ++i) {
      const int rg = m0 + rb + i;
      const float mt = sTM[kn][rb + i];
      float e = ((cg0 <= rg) ? __expf(s0[i] - mt) : 0.f) +
                ((cg1 <= rg) ? __expf(s1[i] - mt) : 0.f);
      eloc[i] = redsum16(e);
    }
    if (lr == 0) {
      #pragma unroll
      for (int i = 0; i < 8; ++i) sPartS[rb + i][waveN] = eloc[i];
    }
    __syncthreads();
    if (tid < 64) sTS[kn][tid] = sPartS[tid][0] + sPartS[tid][1];
    __syncthreads();
  }

  __threadfence();  // make raw score stores visible to all threads' reloads

  // ================= Phase 2: exact row stats =================
  if (tid < 64) {
    float m = NEG_INF;
    for (int kn = 0; kn <= mBlk; ++kn) m = fmaxf(m, sTM[kn][tid]);
    float s = 0.f;
    for (int kn = 0; kn <= mBlk; ++kn)
      s += sTS[kn][tid] * __expf(sTM[kn][tid] - m);
    sMfin[tid] = m;
    sInv[tid]  = 1.0f / s;
  }
  __syncthreads();

  // ================= Phase 3: normalize maps + y = P @ V =================
  v8f y0 = {0.f, 0.f, 0.f, 0.f, 0.f, 0.f, 0.f, 0.f};
  v8f y1 = {0.f, 0.f, 0.f, 0.f, 0.f, 0.f, 0.f, 0.f};
  const int nTiles = T / 64;
  for (int kn = 0; kn < nTiles; ++kn) {
    const int n0t = kn * 64;
    if (kn <= mBlk) {
      // stage V tile transposed to [d][keyRow]
      #pragma unroll
      for (int cidx = 0; cidx < 2; ++cidx) {
        const int chunk = tid + cidx * 256;
        const int c = chunk >> 3, d0 = (chunk & 7) * 8;
        const float* p = vp + (size_t)(n0t + c) * E3 + d0;
        v4f x0 = *(const v4f*)p, x1 = *(const v4f*)(p + 4);
        v8bf t = pack_bf16(x0, x1);
        #pragma unroll
        for (int i = 0; i < 8; ++i) sKV[d0 + i][c] = t[i];
      }
      // P tile: L2-hot reload of own scores -> softmax -> map write + LDS
      for (int j = tid; j < 4096; j += 256) {
        const int r = j >> 6, c = j & 63;
        const int rg = m0 + r, cg = n0t + c;
        const float sv = attZ[(size_t)rg * T + cg];
        const float pv = (cg <= rg) ? __expf(sv - sMfin[r]) * sInv[r] : 0.f;
        attZ[(size_t)rg * T + cg] = pv;
        sP[r][c] = (bf16_t)pv;
      }
      __syncthreads();
      #pragma unroll
      for (int ks = 0; ks < 2; ++ks) {
        v16bf af = *(const v16bf*)&sP[waveM * 16 + lr][ks * 32 + halfSel * 16];
        v16bf bf0 = *(const v16bf*)&sKV[waveN * 32 + lr][ks * 32 + halfSel * 16];
        v16bf bf1 = *(const v16bf*)&sKV[waveN * 32 + 16 + lr][ks * 32 + halfSel * 16];
        y0 = __builtin_amdgcn_wmma_f32_16x16x32_bf16(false, af, false, bf0,
                                                     (short)0, y0, false, false);
        y1 = __builtin_amdgcn_wmma_f32_16x16x32_bf16(false, af, false, bf1,
                                                     (short)0, y1, false, false);
      }
      __syncthreads();
    } else {
      // fully masked tile: zero the map region
      for (int j = tid; j < 4096; j += 256) {
        const int r = j >> 6, c = j & 63;
        attZ[(size_t)(m0 + r) * T + n0t + c] = 0.f;
      }
    }
  }

  // write y block: rows m0+rb+i, head-cols cl0/cl1
  #pragma unroll
  for (int i = 0; i < 8; ++i) {
    yp[(size_t)(m0 + rb + i) * E + cl0] = y0[i];
    yp[(size_t)(m0 + rb + i) * E + cl1] = y1[i];
  }
}

// ---------------------------------------------------------------------------
// Embedding: x[b,t,:] = wte[idx[b,t],:] + wpe[t,:]
// ---------------------------------------------------------------------------
__global__ void embed_kernel(const int* __restrict__ idx,
                             const float* __restrict__ wte,
                             const float* __restrict__ wpe,
                             float* __restrict__ x, int T, int E, int n) {
  int i = blockIdx.x * blockDim.x + threadIdx.x;
  if (i >= n) return;
  int e  = i % E;
  int bt = i / E;
  int t  = bt % T;
  int tok = idx[bt];
  x[i] = wte[(size_t)tok * E + e] + wpe[(size_t)t * E + e];
}

// ---------------------------------------------------------------------------
// LayerNorm (one block per row of E=768)
// ---------------------------------------------------------------------------
__global__ __launch_bounds__(256)
void ln_kernel(const float* __restrict__ x, const float* __restrict__ w,
               const float* __restrict__ b, float* __restrict__ out, int E) {
  __shared__ float rs[256];
  __shared__ float rq[256];
  const int row = blockIdx.x;
  const int tid = threadIdx.x;
  const float* xr = x + (size_t)row * E;
  float s = 0.f, q = 0.f;
  for (int j = tid; j < E; j += 256) { float v = xr[j]; s += v; q += v * v; }
  rs[tid] = s; rq[tid] = q;
  __syncthreads();
  for (int off = 128; off > 0; off >>= 1) {
    if (tid < off) { rs[tid] += rs[tid + off]; rq[tid] += rq[tid + off]; }
    __syncthreads();
  }
  const float mean = rs[0] / (float)E;
  float var = rq[0] / (float)E - mean * mean;
  var = fmaxf(var, 0.0f);
  const float rstd = rsqrtf(var + 1e-5f);
  float* orow = out + (size_t)row * E;
  for (int j = tid; j < E; j += 256)
    orow[j] = (xr[j] - mean) * rstd * w[j] + b[j];
}

// ---------------------------------------------------------------------------
// Exact GELU, elementwise in place
// ---------------------------------------------------------------------------
__global__ void gelu_kernel(float* __restrict__ m, int n) {
  int i = blockIdx.x * blockDim.x + threadIdx.x;
  if (i >= n) return;
  float x = m[i];
  m[i] = 0.5f * x * (1.0f + erff(x * 0.70710678118654752f));
}

// ---------------------------------------------------------------------------
// Host orchestration
// ---------------------------------------------------------------------------
extern "C" void kernel_launch(void* const* d_in, const int* in_sizes, int n_in,
                              void* d_out, int out_size, void* d_ws, size_t ws_size,
                              hipStream_t stream) {
  (void)in_sizes; (void)n_in; (void)out_size; (void)ws_size;
  const int Lc = 6, Bc = 2, Tc = 1024, Ec = 768, Hc = 12, Vc = 50257;
  const int E3 = 3 * Ec, E4 = 4 * Ec;
  const int BT = Bc * Tc;            // 2048
  const float scale = 0.125f;        // 1/sqrt(64)

  const int*   idx  = (const int*)  d_in[0];
  const float* wte  = (const float*)d_in[1];
  const float* wpe  = (const float*)d_in[2];
  const float* ln1w = (const float*)d_in[3];
  const float* ln1b = (const float*)d_in[4];
  const float* attw = (const float*)d_in[5];
  const float* attb = (const float*)d_in[6];
  const float* prjw = (const float*)d_in[7];
  const float* prjb = (const float*)d_in[8];
  const float* ln2w = (const float*)d_in[9];
  const float* ln2b = (const float*)d_in[10];
  const float* fcw  = (const float*)d_in[11];
  const float* fcb  = (const float*)d_in[12];
  const float* fpw  = (const float*)d_in[13];
  const float* fpb  = (const float*)d_in[14];
  const float* lnfw = (const float*)d_in[15];
  const float* lnfb = (const float*)d_in[16];

  float* out = (float*)d_out;
  float* ws  = (float*)d_ws;
  float* x    = ws;                              // [BT,E] residual
  float* h    = x    + (size_t)BT * Ec;          // [BT,E] LN output
  float* qkv  = h    + (size_t)BT * Ec;          // [BT,3E]
  float* y    = qkv  + (size_t)BT * E3;          // [BT,E] attn output
  float* mbuf = y    + (size_t)BT * Ec;          // [BT,4E] MLP hidden

  float* logits = out;                           // [B,V]
  float* attAll = out + (size_t)Bc * Vc;         // [L,B,H,T,T]

  // Embedding
  {
    int n = BT * Ec;
    embed_kernel<<<(n + 255) / 256, 256, 0, stream>>>(idx, wte, wpe, x, Tc, Ec, n);
  }

  for (int l = 0; l < Lc; ++l) {
    const float* aw  = attw + (size_t)l * E3 * Ec;
    const float* ab  = attb + (size_t)l * E3;
    const float* pw  = prjw + (size_t)l * Ec * Ec;
    const float* pb  = prjb + (size_t)l * Ec;
    const float* fw  = fcw  + (size_t)l * E4 * Ec;
    const float* fb  = fcb  + (size_t)l * E4;
    const float* pw2 = fpw  + (size_t)l * Ec * E4;
    const float* pb2 = fpb  + (size_t)l * Ec;
    float* attL = attAll + (size_t)l * Bc * Hc * Tc * Tc;

    // LN1
    ln_kernel<<<BT, 256, 0, stream>>>(x, ln1w + (size_t)l * Ec,
                                      ln1b + (size_t)l * Ec, h, Ec);
    // qkv = h @ attn_w^T + attn_b          [BT,3E]
    {
      dim3 g(E3 / BN, BT / BM, 1);
      gemm_bf16w<true, false, false><<<g, GEMM_THREADS, 0, stream>>>(
          h, aw, ab, qkv, BT, E3, Ec, Ec, Ec, E3);
    }
    // fused attention: scores -> maps (+stats) -> softmax -> y = P@V
    {
      dim3 g(1, Tc / 64, Bc * Hc);
      attn_fused_kernel<<<g, 256, 0, stream>>>(qkv, attL, y, Tc, E3, Ec, Hc, scale);
    }
    // x += y @ proj_w^T + proj_b
    {
      dim3 g(Ec / BN, BT / BM, 1);
      gemm_bf16w<true, true, false><<<g, GEMM_THREADS, 0, stream>>>(
          y, pw, pb, x, BT, Ec, Ec, Ec, Ec, Ec);
    }
    // LN2
    ln_kernel<<<BT, 256, 0, stream>>>(x, ln2w + (size_t)l * Ec,
                                      ln2b + (size_t)l * Ec, h, Ec);
    // m = h @ fc_w^T + fc_b                [BT,4E]
    {
      dim3 g(E4 / BN, BT / BM, 1);
      gemm_bf16w<true, false, false><<<g, GEMM_THREADS, 0, stream>>>(
          h, fw, fb, mbuf, BT, E4, Ec, Ec, Ec, E4);
    }
    // GELU in place
    {
      int n = BT * E4;
      gelu_kernel<<<(n + 255) / 256, 256, 0, stream>>>(mbuf, n);
    }
    // x += m @ fcp_w^T + fcp_b
    {
      dim3 g(Ec / BN, BT / BM, 1);
      gemm_bf16w<true, true, false><<<g, GEMM_THREADS, 0, stream>>>(
          mbuf, pw2, pb2, x, BT, Ec, E4, E4, E4, Ec);
    }
  }

  // Final LN
  ln_kernel<<<BT, 256, 0, stream>>>(x, lnfw, lnfb, h, Ec);

  // logits = lnf(x)[:, -1, :] @ wte^T   (M = B = 2 rows, stride T*E; ragged)
  {
    dim3 g((Vc + BN - 1) / BN, 1, 1);
    gemm_bf16w<false, false, true><<<g, GEMM_THREADS, 0, stream>>>(
        h + (size_t)(Tc - 1) * Ec, wte, nullptr, logits, Bc, Vc, Ec,
        (long long)Tc * Ec, (long long)Ec, (long long)Vc);
  }
}
